// PDIScore_12644383719683
// MI455X (gfx1250) — matrix operand order repaired and checked by reading, not compiled, not run on hardware
//
#include <hip/hip_runtime.h>
#include <hip/hip_bf16.h>

#define BSZ 8
#define NL 128
#define NT 512
#define CIN 128
#define HID 128
#define NG 10
#define ATOMS 24
#define NEDGES 4096

typedef __attribute__((ext_vector_type(16))) _Float16 v16h;
typedef __attribute__((ext_vector_type(8)))  float    v8f;
typedef __attribute__((ext_vector_type(8)))  unsigned int v8u;

__device__ __forceinline__ float elu_f(float x) {
    return x > 0.f ? x : (__expf(x) - 1.f);
}

// ---------------------------------------------------------------------------
// Fold BatchNorm + b1 into per-channel scale/shift:
//   h = (lin + b1 - mean) * gamma*rsqrt(var+eps) + beta = lin*scale + shift
// ---------------------------------------------------------------------------
__global__ __launch_bounds__(128) void prep_affine(
    const float* __restrict__ b1, const float* __restrict__ gamma,
    const float* __restrict__ beta, const float* __restrict__ mean,
    const float* __restrict__ var, float* __restrict__ scale,
    float* __restrict__ shift)
{
    int i = threadIdx.x;
    float s = gamma[i] * rsqrtf(var[i] + 1e-5f);
    scale[i] = s;
    shift[i] = beta[i] + (b1[i] - mean[i]) * s;
}

// ---------------------------------------------------------------------------
// Pack [W_pi | W_sigma | W_mu | 0 0] (128 x 32, f32) into WMMA B-fragment
// lane order.  B layout (16-bit, 32x16 per k-step): column n = lane&15,
// lanes 0-15 hold K=0..7,16..23, lanes 16-31 hold K=8..15,24..31, VGPR v = pair (2v,2v+1).
// Flat layout: Bfrag[tile][kstep][lane][element], element e -> K offset 8*(lane>>4) + (e&7) + 16*(e>>3).
// ---------------------------------------------------------------------------
__global__ __launch_bounds__(256) void prep_bfrag(
    const float* __restrict__ Wpi, const float* __restrict__ Wsig,
    const float* __restrict__ Wmu, _Float16* __restrict__ Bfrag)
{
    int idx = blockIdx.x * 256 + threadIdx.x;      // 0..4095
    int e = idx & 15;
    int l = (idx >> 4) & 31;
    int s = (idx >> 9) & 3;
    int t = idx >> 11;
    int n = t * 16 + (l & 15);
    int k = 32 * s + ((e >> 3) << 4) + ((l >> 4) << 3) + (e & 7);
    float val = 0.f;
    if (n < 10)       val = Wpi[k * NG + n];
    else if (n < 20)  val = Wsig[k * NG + (n - 10)];
    else if (n < 30)  val = Wmu[k * NG + (n - 20)];
    Bfrag[idx] = (_Float16)val;
}

// ---------------------------------------------------------------------------
// A_l = h_l_x @ W1[0:128,:]  (rows 0..1023)
// A_t = h_t_x @ W1[128:256,:] (rows 0..4095)
// ---------------------------------------------------------------------------
__global__ __launch_bounds__(128) void prep_A(
    const float* __restrict__ hl, const float* __restrict__ ht,
    const float* __restrict__ W1, float* __restrict__ A_l,
    float* __restrict__ A_t)
{
    __shared__ float x_sh[CIN];
    int row = blockIdx.x;
    const float* x; const float* W; float* out;
    if (row < BSZ * NL) {
        x = hl + (size_t)row * CIN;  W = W1;              out = A_l + (size_t)row * HID;
    } else {
        int r = row - BSZ * NL;
        x = ht + (size_t)r * CIN;    W = W1 + CIN * HID;  out = A_t + (size_t)r * HID;
    }
    int j = threadIdx.x;
    x_sh[j] = x[j];
    __syncthreads();
    float acc = 0.f;
#pragma unroll 8
    for (int k = 0; k < CIN; ++k) acc += x_sh[k] * W[k * HID + j];
    out[j] = acc;
}

// ---------------------------------------------------------------------------
// Fused pair kernel: one block per (b,l).  4 waves, each wave owns a 16-row
// t-tile per iteration: build h = ELU((A_l+A_t)*scale+shift) in f16 LDS,
// WMMA 16x16x32 f16 against the 128x32 head matrix (2 N-tiles x 4 K-steps),
// stage D in LDS, epilogue softmax/ELU + stores.
// ---------------------------------------------------------------------------
__global__ __launch_bounds__(128) void fused_pair_kernel(
    const float* __restrict__ A_l, const float* __restrict__ A_t,
    const float* __restrict__ scale, const float* __restrict__ shift,
    const _Float16* __restrict__ Bfrag,
    const float* __restrict__ b_pi, const float* __restrict__ b_sigma,
    const float* __restrict__ b_mu,
    float* __restrict__ out_pi, float* __restrict__ out_sigma,
    float* __restrict__ out_mu)
{
    __shared__ float a_sh[HID];
    __shared__ float sc_sh[HID];
    __shared__ float sh_sh[HID];
    __shared__ float bias_sh[32];
    __shared__ _Float16 htile[4][16][HID + 8];   // +8 halves pad: conflict-free dword reads
    __shared__ float dstage[4][16][33];          // +1 pad

    const int tid  = threadIdx.x;
    const int wave = tid >> 5;
    const int lane = tid & 31;
    const int bl   = blockIdx.x;                 // b*NL + l
    const int b    = bl >> 7;

    a_sh[tid]  = A_l[(size_t)bl * HID + tid];
    sc_sh[tid] = scale[tid];
    sh_sh[tid] = shift[tid];
    if (tid < 32) {
        float bv = 0.f;
        if (tid < 10)       bv = b_pi[tid];
        else if (tid < 20)  bv = b_sigma[tid - 10];
        else if (tid < 30)  bv = b_mu[tid - 20];
        bias_sh[tid] = bv;
    }
    __syncthreads();

    // Preload the 8 B-fragments (2 N-tiles x 4 K-steps), resident in VGPRs.
    v16h bfr[2][4];
    const v16h* bf = reinterpret_cast<const v16h*>(Bfrag);
#pragma unroll
    for (int t = 0; t < 2; ++t)
#pragma unroll
        for (int s = 0; s < 4; ++s)
            bfr[t][s] = bf[(t * 4 + s) * 32 + lane];

    const int m     = lane & 15;            // A row / D staging helpers
    const int kbase = (lane >> 4) << 3;     // A layout: lanes 16-31 start at K=8
    const int rbase = (lane >> 4) << 3;     // D layout: lanes 16-31 hold M=v+8
    const int n     = lane & 15;

    for (int iter = 0; iter < 8; ++iter) {
        const int t0 = (iter * 4 + wave) * 16;

        // ---- build ELU'd h tile (16 t-rows x 128 ch) in f16 LDS ----
        for (int idx = lane; idx < 16 * HID; idx += 32) {
            int r = idx >> 7;
            int c = idx & (HID - 1);
            float v = (a_sh[c] + A_t[((size_t)(b * NT + t0 + r)) * HID + c]) * sc_sh[c] + sh_sh[c];
            htile[wave][r][c] = (_Float16)elu_f(v);
        }
        __syncthreads();

        // ---- 8x WMMA f32_16x16x32_f16 ----
        v8f acc0 = {};
        v8f acc1 = {};
#pragma unroll
        for (int s = 0; s < 4; ++s) {
            union { v8u u; v16h h; } af;
#pragma unroll
            for (int vv = 0; vv < 8; ++vv) {
                int k = 32 * s + kbase + ((vv >> 2) << 4) + ((vv & 3) << 1);
                af.u[vv] = *reinterpret_cast<const unsigned int*>(&htile[wave][m][k]);
            }
            acc0 = __builtin_amdgcn_wmma_f32_16x16x32_f16(
                false, af.h, false, bfr[0][s], (short)0, acc0, false, false);
            acc1 = __builtin_amdgcn_wmma_f32_16x16x32_f16(
                false, af.h, false, bfr[1][s], (short)0, acc1, false, false);
        }

        // ---- stage D (C/D layout: VGPR v -> row v + rbase, col = lane&15) ----
#pragma unroll
        for (int vv = 0; vv < 8; ++vv) {
            dstage[wave][rbase + vv][n]      = acc0[vv];
            dstage[wave][rbase + vv][n + 16] = acc1[vv];
        }
        __syncthreads();

        // ---- epilogue: one lane per t-row ----
        if (lane < 16) {
            int r = lane;
            int row = bl * NT + t0 + r;
            float z[30];
#pragma unroll
            for (int j = 0; j < 30; ++j) z[j] = dstage[wave][r][j] + bias_sh[j];
            // softmax over pi
            float mx = z[0];
#pragma unroll
            for (int j = 1; j < NG; ++j) mx = fmaxf(mx, z[j]);
            float e[NG]; float sum = 0.f;
#pragma unroll
            for (int j = 0; j < NG; ++j) { e[j] = __expf(z[j] - mx); sum += e[j]; }
            float inv = 1.f / sum;
#pragma unroll
            for (int j = 0; j < NG; ++j) out_pi[(size_t)row * NG + j] = e[j] * inv;
#pragma unroll
            for (int j = 0; j < NG; ++j)
                out_sigma[(size_t)row * NG + j] = elu_f(z[10 + j]) + 1.1f;
#pragma unroll
            for (int j = 0; j < NG; ++j)
                out_mu[(size_t)row * NG + j] = elu_f(z[20 + j]) + 1.0f;
        }
        __syncthreads();
    }
}

// ---------------------------------------------------------------------------
// dist[b,l,t] = min_a sqrt(|p-q|^2) with sqrt(neg)->10000 ; also C_batch = b
// ---------------------------------------------------------------------------
__global__ __launch_bounds__(256) void dist_kernel(
    const float* __restrict__ lpos, const float* __restrict__ tpos,
    float* __restrict__ out_dist, float* __restrict__ out_cb)
{
    int idx = blockIdx.x * 256 + threadIdx.x;    // (b*NL+l)*NT + t
    int t  = idx & (NT - 1);
    int bl = idx >> 9;
    int b  = idx >> 16;
    float px = lpos[bl * 3 + 0];
    float py = lpos[bl * 3 + 1];
    float pz = lpos[bl * 3 + 2];
    float p2 = px * px + py * py + pz * pz;
    const float* q = tpos + ((size_t)(b * NT + t) * ATOMS) * 3;
    float best = 1e30f;
#pragma unroll 4
    for (int a = 0; a < ATOMS; ++a) {
        float qx = q[a * 3 + 0], qy = q[a * 3 + 1], qz = q[a * 3 + 2];
        float d2 = p2 + qx * qx + qy * qy + qz * qz
                 - 2.f * (px * qx + py * qy + pz * qz);
        float d = d2 < 0.f ? 10000.0f : sqrtf(d2);
        best = fminf(best, d);
    }
    out_dist[idx] = best;
    out_cb[idx]   = (float)b;
}

// ---------------------------------------------------------------------------
// nuc_types = x_l_flat @ W_nuc + b_nuc   ([1024,128] x [128,12])
// ---------------------------------------------------------------------------
__global__ __launch_bounds__(256) void nuc_kernel(
    const float* __restrict__ x, const float* __restrict__ W,
    const float* __restrict__ bias, float* __restrict__ out)
{
    int idx = blockIdx.x * 256 + threadIdx.x;    // 0..12287
    int nn = idx % 12, r = idx / 12;
    float acc = bias[nn];
#pragma unroll 8
    for (int k = 0; k < CIN; ++k) acc += x[(size_t)r * CIN + k] * W[k * 12 + nn];
    out[idx] = acc;
}

// ---------------------------------------------------------------------------
// bond_types = [x[e0]|x[e1]] @ W_bond + b_bond   ([4096,256] x [256,4])
// ---------------------------------------------------------------------------
__global__ __launch_bounds__(256) void bond_kernel(
    const float* __restrict__ x, const int* __restrict__ eidx,
    const float* __restrict__ W, const float* __restrict__ bias,
    float* __restrict__ out)
{
    int idx = blockIdx.x * 256 + threadIdx.x;    // 0..16383
    int nn = idx & 3, e = idx >> 2;
    int e0 = eidx[e];
    int e1 = eidx[NEDGES + e];
    float acc = bias[nn];
#pragma unroll 8
    for (int k = 0; k < CIN; ++k) acc += x[(size_t)e0 * CIN + k] * W[k * 4 + nn];
#pragma unroll 8
    for (int k = 0; k < CIN; ++k) acc += x[(size_t)e1 * CIN + k] * W[(CIN + k) * 4 + nn];
    out[idx] = acc;
}

extern "C" void kernel_launch(void* const* d_in, const int* in_sizes, int n_in,
                              void* d_out, int out_size, void* d_ws, size_t ws_size,
                              hipStream_t stream) {
    const float* h_l_x   = (const float*)d_in[0];
    const float* h_t_x   = (const float*)d_in[1];
    const float* h_l_pos = (const float*)d_in[4];
    const float* h_t_pos = (const float*)d_in[5];
    const float* x_l     = (const float*)d_in[6];
    const int*   eidx    = (const int*)d_in[7];
    const float* W1      = (const float*)d_in[8];
    const float* b1      = (const float*)d_in[9];
    const float* gamma   = (const float*)d_in[10];
    const float* beta    = (const float*)d_in[11];
    const float* mean    = (const float*)d_in[12];
    const float* var     = (const float*)d_in[13];
    const float* W_pi    = (const float*)d_in[14];
    const float* b_pi    = (const float*)d_in[15];
    const float* W_sig   = (const float*)d_in[16];
    const float* b_sig   = (const float*)d_in[17];
    const float* W_mu    = (const float*)d_in[18];
    const float* b_mu    = (const float*)d_in[19];
    const float* W_nuc   = (const float*)d_in[20];
    const float* b_nuc   = (const float*)d_in[21];
    const float* W_bond  = (const float*)d_in[22];
    const float* b_bond  = (const float*)d_in[23];

    // workspace layout (floats)
    float* ws     = (float*)d_ws;
    float* scale  = ws;                    // 128
    float* shift  = ws + 128;              // 128
    float* A_l    = ws + 256;              // 1024*128
    float* A_t    = A_l + BSZ * NL * HID;  // 4096*128
    _Float16* Bfrag = (_Float16*)(A_t + BSZ * NT * HID);   // 4096 halves

    // output layout (flat concat in reference return order)
    const size_t M = (size_t)BSZ * NL * NT;      // 524288
    float* out       = (float*)d_out;
    float* out_pi    = out;
    float* out_sigma = out + M * NG;
    float* out_mu    = out + 2 * M * NG;
    float* out_dist  = out + 3 * M * NG;
    float* out_nuc   = out_dist + M;
    float* out_bond  = out_nuc + BSZ * NL * 12;
    float* out_cb    = out_bond + NEDGES * 4;

    prep_affine<<<1, 128, 0, stream>>>(b1, gamma, beta, mean, var, scale, shift);
    prep_bfrag<<<16, 256, 0, stream>>>(W_pi, W_sig, W_mu, Bfrag);
    prep_A<<<BSZ * NL + BSZ * NT, 128, 0, stream>>>(h_l_x, h_t_x, W1, A_l, A_t);

    fused_pair_kernel<<<BSZ * NL, 128, 0, stream>>>(
        A_l, A_t, scale, shift, Bfrag, b_pi, b_sig, b_mu,
        out_pi, out_sigma, out_mu);

    dist_kernel<<<(int)(M / 256), 256, 0, stream>>>(h_l_pos, h_t_pos, out_dist, out_cb);
    nuc_kernel<<<48, 256, 0, stream>>>(x_l, W_nuc, b_nuc, out_nuc);
    bond_kernel<<<64, 256, 0, stream>>>(x_l, eidx, W_bond, b_bond, out_bond);
}
